// MinSumBPDecoder_22677427323114
// MI455X (gfx1250) — 8-bit matrix operand order repaired and checked by reading, compiled, not run on hardware
//
#include <hip/hip_runtime.h>
#include <hip/hip_bf16.h>
#include <cstdint>

#define M_CHK   4096
#define N_VAR   8192
#define MAXITER 10
#define ALPHA_C 0.8f
#define CLAMP_C 20.0f
#define PAD_C   1000000.0f

typedef __attribute__((ext_vector_type(8))) int v8i;

__global__ void k_pcm_to_u8(const float* __restrict__ pcm,
                            uint8_t* __restrict__ pcm8, long long n) {
  long long g = (long long)blockIdx.x * blockDim.x + threadIdx.x;
  if (g < n) pcm8[g] = (uint8_t)(pcm[g] != 0.0f);
}

__global__ void k_prep(const float* __restrict__ llr,
                       const float* __restrict__ syndrome,
                       float* __restrict__ llrT,
                       float* __restrict__ ssignT,
                       int B, int M, int N) {
  int g = blockIdx.x * blockDim.x + threadIdx.x;
  int nb = N * B;
  if (g < nb) {
    int v = g / B, b = g % B;
    llrT[g] = llr[(size_t)b * N + v];
  } else {
    int h = g - nb;
    if (h < M * B) {
      int c = h / B, b = h % B;
      ssignT[h] = 1.0f - 2.0f * syndrome[(size_t)b * M + c];
    }
  }
}

__global__ void k_var_total(const int* __restrict__ var_adj,
                            const float* __restrict__ var_mask,
                            const float* __restrict__ ctv,
                            float* __restrict__ vtot,
                            int B, int max_dv, int N) {
  int g = blockIdx.x * blockDim.x + threadIdx.x;
  if (g >= N * B) return;
  int v = g / B, b = g % B;
  float s = 0.0f;
  for (int j = 0; j < max_dv; ++j) {
    if (var_mask[v * max_dv + j] == 0.0f) break;
    int e = var_adj[v * max_dv + j];
    s += ctv[(size_t)e * B + b];
  }
  vtot[g] = s;
}

__global__ void k_check_update(const int* __restrict__ check_adj,
                               const float* __restrict__ check_mask,
                               const int* __restrict__ var_idx,
                               const float* __restrict__ llrT,
                               const float* __restrict__ ssignT,
                               const float* __restrict__ vtot,
                               float* __restrict__ ctv,
                               int B, int max_dc, int M) {
  int g = blockIdx.x * blockDim.x + threadIdx.x;
  if (g >= M * B) return;
  int c = g / B, b = g % B;

  float sgnprod = 1.0f;
  float min1 = PAD_C, min2 = PAD_C;
  int dc = 0;
  for (int j = 0; j < max_dc; ++j) {
    if (check_mask[c * max_dc + j] == 0.0f) break;
    int e = check_adj[c * max_dc + j];
    int v = var_idx[e];
    float t = llrT[(size_t)v * B + b] + vtot[(size_t)v * B + b]
            - ctv[(size_t)e * B + b];
    t = fminf(fmaxf(t, -CLAMP_C), CLAMP_C);
    if (t < 0.0f) sgnprod = -sgnprod;
    float a = fabsf(t);
    if (a < min1) { min2 = min1; min1 = a; }
    else if (a < min2) { min2 = a; }
    dc = j + 1;
  }
  float total_sign = ssignT[(size_t)c * B + b] * sgnprod;

  for (int j = 0; j < dc; ++j) {
    int e = check_adj[c * max_dc + j];
    int v = var_idx[e];
    float t = llrT[(size_t)v * B + b] + vtot[(size_t)v * B + b]
            - ctv[(size_t)e * B + b];
    t = fminf(fmaxf(t, -CLAMP_C), CLAMP_C);
    float s  = (t >= 0.0f) ? 1.0f : -1.0f;
    float a  = fabsf(t);
    float em = (fabsf(a - min1) < 1e-9f) ? min2 : min1;
    ctv[(size_t)e * B + b] = ALPHA_C * total_sign * s * em;
  }
}

__global__ void k_finalize(const float* __restrict__ llr,
                           const float* __restrict__ vtot,
                           float* __restrict__ marg,
                           int* __restrict__ hard,
                           uint8_t* __restrict__ hard8,
                           int B, int N) {
  int g = blockIdx.x * blockDim.x + threadIdx.x;
  if (g >= B * N) return;
  int b = g / N, v = g % N;
  float total = llr[g] + vtot[(size_t)v * B + b];
  float m = 1.0f / (1.0f + expf(total));
  int hd = (m > 0.5f) ? 1 : 0;
  marg[g]  = m;
  hard[g]  = hd;
  hard8[g] = (uint8_t)hd;
}

__global__ void k_syndrome_wmma(const uint8_t* __restrict__ hard8,
                                const uint8_t* __restrict__ pcm8,
                                const float* __restrict__ syndrome,
                                unsigned* __restrict__ mis,
                                int B, int N, int M) {
  int wave = (blockIdx.x * blockDim.x + threadIdx.x) >> 5;
  int lane = threadIdx.x & 31;
  int tilesM = M >> 4;
  int tb = wave / tilesM;
  int tc = wave % tilesM;
  if (tb >= (B >> 4)) return;

  int r      = lane & 15;
  int hiA    = (lane >> 4) ? 8  : 0;
  int hiB    = (lane >> 4) ? 8  : 0;
  const uint8_t* arow = hard8 + (size_t)(tb * 16 + r) * N;
  const uint8_t* brow = pcm8  + (size_t)(tc * 16 + r) * N;

  const int aoff[8] = {0, 4, 16, 20, 32, 36, 48, 52};
  const int boff[8] = {0, 4, 16, 20, 32, 36, 48, 52};

  v8i acc = {0, 0, 0, 0, 0, 0, 0, 0};
  for (int k0 = 0; k0 < N; k0 += 64) {
    __builtin_prefetch(arow + k0 + 256, 0, 0);
    __builtin_prefetch(brow + k0 + 256, 0, 0);
    v8i a, b;
#pragma unroll
    for (int i = 0; i < 8; ++i) {
      a[i] = *(const int*)(arow + k0 + hiA + aoff[i]);
      b[i] = *(const int*)(brow + k0 + hiB + boff[i]);
    }
    acc = __builtin_amdgcn_wmma_i32_16x16x64_iu8(false, a, false, b, acc,
                                                 false, false);
  }

  int n     = lane & 15;
  int mbase = (lane >> 4) ? 8 : 0;
  int cg    = tc * 16 + n;
#pragma unroll
  for (int i = 0; i < 8; ++i) {
    int bg = tb * 16 + mbase + i;
    float parity = (float)(acc[i] & 1);
    if (parity != syndrome[(size_t)bg * M + cg])
      atomicAdd(&mis[bg], 1u);
  }
}

__global__ void k_converged(const unsigned* __restrict__ mis,
                            float* __restrict__ conv, int B) {
  int b = blockIdx.x * blockDim.x + threadIdx.x;
  if (b < B) conv[b] = (mis[b] == 0u) ? 1.0f : 0.0f;
}

extern "C" void kernel_launch(void* const* d_in, const int* in_sizes, int n_in,
                              void* d_out, int out_size, void* d_ws, size_t ws_size,
                              hipStream_t stream) {
  const float* syndrome   = (const float*)d_in[0];
  const float* llr        = (const float*)d_in[1];
  const int*   var_adj    = (const int*)  d_in[2];
  const float* var_mask   = (const float*)d_in[3];
  const int*   check_adj  = (const int*)  d_in[4];
  const float* check_mask = (const float*)d_in[5];
  const int*   var_idx    = (const int*)  d_in[6];
  const float* pcm        = (const float*)d_in[7];

  const int M = M_CHK, N = N_VAR;
  const int B      = in_sizes[0] / M;
  const int E      = in_sizes[6];
  const int max_dv = in_sizes[2] / N;
  const int max_dc = in_sizes[4] / M;

  char* ws = (char*)d_ws;
  size_t off = 0;
  auto carve = [&](size_t bytes) -> char* {
    char* p = ws + off;
    off += (bytes + 255) & ~(size_t)255;
    return p;
  };
  float*    ctv    = (float*)   carve((size_t)E * B * 4);
  float*    vtot   = (float*)   carve((size_t)N * B * 4);
  float*    llrT   = (float*)   carve((size_t)N * B * 4);
  float*    ssignT = (float*)   carve((size_t)M * B * 4);
  uint8_t*  pcm8   = (uint8_t*) carve((size_t)M * N);
  uint8_t*  hard8  = (uint8_t*) carve((size_t)B * N);
  unsigned* mis    = (unsigned*)carve((size_t)B * 4);
  (void)ws_size; (void)n_in;

  hipMemsetAsync(ctv, 0, (size_t)E * B * 4, stream);
  hipMemsetAsync(mis, 0, (size_t)B * 4, stream);

  const int T = 256;
  long long pcmN = (long long)M * N;
  k_pcm_to_u8<<<(unsigned)((pcmN + T - 1) / T), T, 0, stream>>>(pcm, pcm8, pcmN);

  int prepN = N * B + M * B;
  k_prep<<<(prepN + T - 1) / T, T, 0, stream>>>(llr, syndrome, llrT, ssignT, B, M, N);

  for (int it = 0; it < MAXITER; ++it) {
    k_var_total<<<(N * B + T - 1) / T, T, 0, stream>>>(var_adj, var_mask, ctv,
                                                       vtot, B, max_dv, N);
    k_check_update<<<(M * B + T - 1) / T, T, 0, stream>>>(check_adj, check_mask,
                                                          var_idx, llrT, ssignT,
                                                          vtot, ctv, B, max_dc, M);
  }
  k_var_total<<<(N * B + T - 1) / T, T, 0, stream>>>(var_adj, var_mask, ctv,
                                                     vtot, B, max_dv, N);

  float* marg = (float*)d_out;
  int*   hard = (int*)((float*)d_out + (size_t)B * N);
  float* conv = (float*)d_out + 2 * (size_t)B * N;

  k_finalize<<<(B * N + T - 1) / T, T, 0, stream>>>(llr, vtot, marg, hard,
                                                    hard8, B, N);

  int waves = (B / 16) * (M / 16);
  k_syndrome_wmma<<<(waves * 32 + T - 1) / T, T, 0, stream>>>(hard8, pcm8,
                                                              syndrome, mis,
                                                              B, N, M);
  k_converged<<<(B + T - 1) / T, T, 0, stream>>>(mis, conv, B);
}
